// SourceQueryEvidenceExtractor_48387101556848
// MI455X (gfx1250) — matrix operand order repaired and checked by reading, not compiled, not run on hardware
//
#include <hip/hip_runtime.h>

// ---------------------------------------------------------------------------
// CDNA5 (gfx1250) bf16 WMMA types and helpers
// ---------------------------------------------------------------------------
typedef __attribute__((ext_vector_type(16))) __bf16 v16bf;
typedef __attribute__((ext_vector_type(8)))  __bf16 v8bf;
typedef __attribute__((ext_vector_type(8)))  float  v8f;
typedef __attribute__((ext_vector_type(4)))  int    v4i;

#if defined(__has_builtin)
#if __has_builtin(__builtin_amdgcn_global_load_async_to_lds_b128) && \
    __has_builtin(__builtin_amdgcn_s_wait_asynccnt)
#define USE_ASYNC_COPY 1
#endif
#endif
#ifndef USE_ASYNC_COPY
#define USE_ASYNC_COPY 0
#endif

__device__ __forceinline__ v8bf ldv8(const __bf16* p) {
    return *reinterpret_cast<const v8bf*>(p);
}
__device__ __forceinline__ v16bf mk16(v8bf lo, v8bf hi) {
    v16bf r;
#pragma unroll
    for (int i = 0; i < 8; i++) { r[i] = lo[i]; r[i + 8] = hi[i]; }
    return r;
}
__device__ __forceinline__ v8f wmma_bf16(v16bf a, v16bf b, v8f c) {
    return __builtin_amdgcn_wmma_f32_16x16x32_bf16(false, a, false, b, (short)0, c, false, false);
}
// A 16x32 fragment: lane m = lane&15; elems 0-7 -> K = k+half*8..; 8-15 -> K = k+16+half*8..
__device__ __forceinline__ v16bf ldA(const __bf16* arow, int k, int half) {
    return mk16(ldv8(arow + k + half * 8), ldv8(arow + k + 16 + half * 8));
}

// 16-byte global->LDS copy: async on CDNA5 (GLOBAL_LOAD_ASYNC_TO_LDS_B128),
// register bounce otherwise.  Builtin signature: (v4i AS1* src, v4i AS3* dst, imm off, imm cpol)
__device__ __forceinline__ void copy16(const __bf16* g, __bf16* l) {
#if USE_ASYNC_COPY
    __builtin_amdgcn_global_load_async_to_lds_b128(
        (__attribute__((address_space(1))) v4i*)(const_cast<__bf16*>(g)),
        (__attribute__((address_space(3))) v4i*)(l), 0, 0);
#else
    *reinterpret_cast<v8bf*>(l) = *reinterpret_cast<const v8bf*>(g);
#endif
}
// Wait for all but `pending` of this wave's async fills, then block barrier.
__device__ __forceinline__ void copy_fence(int hasPending) {
#if USE_ASYNC_COPY
    if (hasPending) __builtin_amdgcn_s_wait_asynccnt(1);
    else            __builtin_amdgcn_s_wait_asynccnt(0);
#endif
    __syncthreads();
}
__device__ __forceinline__ void copy_wait_all() {
#if USE_ASYNC_COPY
    __builtin_amdgcn_s_wait_asynccnt(0);
#endif
    __syncthreads();
}

static constexpr int Bn = 4, Cc = 256, Hh = 128, Ww = 128, HW = 128 * 128;
static constexpr int Qd = 256, Ee = 192, Ss = 4;

// ---------------------------------------------------------------------------
// Prep kernels
// ---------------------------------------------------------------------------
__global__ void cvt_f32_bf16(const float* __restrict__ src, __bf16* __restrict__ dst, int n) {
    int i = blockIdx.x * 256 + threadIdx.x;
    if (i < n) dst[i] = (__bf16)src[i];
}

// out_w (O,I,3,3) f32  ->  wconv[tap][o][i] bf16   (tap = kh*3+kw)
__global__ void prep_wconv(const float* __restrict__ w, __bf16* __restrict__ wc, int n) {
    int i = blockIdx.x * 256 + threadIdx.x;
    if (i >= n) return;
    int ei  = i % Ee;
    int o   = (i / Ee) % Ee;
    int tap = i / (Ee * Ee);
    int kh = tap / 3, kw = tap % 3;
    wc[i] = (__bf16)w[((o * Ee + ei) * 3 + kh) * 3 + kw];
}

__global__ void prep_bn(const float* g1, const float* b1, const float* m1, const float* v1,
                        const float* g2, const float* b2, const float* m2, const float* v2,
                        float* s1, float* h1, float* s2, float* h2) {
    int e = threadIdx.x;
    if (e >= Ee) return;
    float sc1 = g1[e] * rsqrtf(v1[e] + 1e-5f);
    s1[e] = sc1; h1[e] = b1[e] - m1[e] * sc1;
    float sc2 = g2[e] * rsqrtf(v2[e] + 1e-5f);
    s2[e] = sc2; h2[e] = b2[e] - m2[e] * sc2;
}

// memory (B,C,H,W) f32 -> mem_t (B,HW,C) bf16 via LDS tiled transpose
__global__ __launch_bounds__(256) void transpose_mem(const float* __restrict__ mem,
                                                     __bf16* __restrict__ memt) {
    __shared__ float tile[32][33];
    int b  = blockIdx.z;
    int c0 = blockIdx.y * 32, p0 = blockIdx.x * 32;
    int tx = threadIdx.x & 31, ty = threadIdx.x >> 5;
#pragma unroll
    for (int i = 0; i < 4; i++) {
        int c = c0 + ty + i * 8;
        tile[ty + i * 8][tx] = mem[((size_t)b * Cc + c) * HW + p0 + tx];
    }
    __syncthreads();
#pragma unroll
    for (int i = 0; i < 4; i++) {
        int p = p0 + ty + i * 8;
        memt[((size_t)b * HW + p) * Cc + c0 + tx] = (__bf16)tile[tx][ty + i * 8];
    }
}

// ---------------------------------------------------------------------------
// Batched GEMM: out[b][n][m] = act( sum_k Wm[m][k] * X[b][n][k] + bias )
//   Wm: (M,K) bf16 ; X: (B,N,K) bf16 ; M multiple of 32, K multiple of 32
// Block = 256 threads (8 waves) sharing one 64-wide N strip.
//   - B panel (64 x 32) double-buffered in LDS, filled with async copies
//   - each wave: 32(M) x 64(N) tile, A fragments double-buffered in regs
//   mode 0: out bf16, +p0[m] if p0          (mt / k / v projections)
//   mode 1: out f32, silu(y*p0[m] + p1[m])  (evidence, BN1 folded)
//   mode 2: out f32, sigmoid(y + p0[m])     (gate)
// ---------------------------------------------------------------------------
__global__ __launch_bounds__(256) void gemm_bf16_wmma(
    const __bf16* __restrict__ Wm, const __bf16* __restrict__ X,
    void* __restrict__ outp, const float* __restrict__ p0,
    const float* __restrict__ p1, int M, int N, int K, int mode)
{
    __shared__ __bf16 Bs[2][64][40];   // 32 K-elems used + 8 pad (80B row: no bank conflicts)
    const int tid  = threadIdx.x;
    const int lane = tid & 31;
    const int wave = tid >> 5;
    const int half = lane >> 4;
    const int nn   = lane & 15;
    const int batch = blockIdx.z;
    const long n0 = (long)blockIdx.x * 64;
    const int mwaves = M >> 5;
    const bool active = wave < mwaves;
    const int m0 = (active ? wave : 0) << 5;

    auto fillB = [&](int buf, int k) {
        // 64 rows x 32 K-elems = 256 x 16B chunks; one per thread
        int n = tid >> 2, q = tid & 3;
        copy16(X + ((size_t)batch * N + n0 + n) * K + k + q * 8, &Bs[buf][n][q * 8]);
    };

    const __bf16* arow0 = Wm + (size_t)(m0 + nn) * K;
    const __bf16* arow1 = Wm + (size_t)(m0 + 16 + nn) * K;

    fillB(0, 0);
    v16bf a0 = ldA(arow0, 0, half);
    v16bf a1 = ldA(arow1, 0, half);

    v8f acc[8] = {};
    const int nk = K >> 5;
    for (int ks = 0; ks < nk; ks++) {
        const int cur = ks & 1;
        const bool hasNext = (ks + 1 < nk);
        if (hasNext) fillB(cur ^ 1, (ks + 1) << 5);   // async fill of other buffer
        copy_fence(hasNext ? 1 : 0);                  // current panel ready, all waves synced
        v16bf na0 = a0, na1 = a1;
        if (hasNext) {                                 // prefetch next A frags (in flight over WMMAs)
            na0 = ldA(arow0, (ks + 1) << 5, half);
            na1 = ldA(arow1, (ks + 1) << 5, half);
        }
#pragma unroll
        for (int j = 0; j < 4; j++) {
            // B 32x16 fragment from LDS: row n = j*16+nn, elems 0-7 -> K = half*8 + i, 8-15 -> K = 16 + half*8 + i
            v16bf bf = mk16(ldv8(&Bs[cur][j * 16 + nn][half * 8]),
                            ldv8(&Bs[cur][j * 16 + nn][16 + half * 8]));
            acc[j]     = wmma_bf16(a0, bf, acc[j]);
            acc[4 + j] = wmma_bf16(a1, bf, acc[4 + j]);
        }
        a0 = na0; a1 = na1;
        __syncthreads();                               // reads done before this buffer is refilled
    }

    if (!active) return;
#pragma unroll
    for (int mt = 0; mt < 2; mt++) {
#pragma unroll
        for (int j = 0; j < 4; j++) {
            long n = n0 + j * 16 + nn;
            size_t obase = ((size_t)batch * N + n) * M;
#pragma unroll
            for (int r = 0; r < 8; r++) {
                int m = m0 + mt * 16 + r + half * 8;   // D: VGPR r -> M = r + 8*half
                float y = acc[mt * 4 + j][r];
                if (mode == 0) {
                    if (p0) y += p0[m];
                    ((__bf16*)outp)[obase + m] = (__bf16)y;
                } else if (mode == 1) {
                    y = y * p0[m] + p1[m];
                    y = y / (1.f + __expf(-y));        // SiLU
                    ((float*)outp)[obase + m] = y;
                } else {
                    y = 1.f / (1.f + __expf(-(y + p0[m])));
                    ((float*)outp)[obase + m] = y;
                }
            }
        }
    }
}

// ---------------------------------------------------------------------------
// Attention path (S=4 -> tiny, plain VALU)
// ---------------------------------------------------------------------------
__global__ __launch_bounds__(256) void qproj_kernel(const float* __restrict__ sq,
                                                    const float* __restrict__ w,
                                                    const float* __restrict__ bias,
                                                    float* __restrict__ q) {
    int t = blockIdx.x, j = threadIdx.x;
    float acc = bias[j];
    const float* s = sq + (size_t)t * Qd;
    const float* wr = w + (size_t)j * Qd;
    for (int c = 0; c < Qd; c++) acc += s[c] * wr[c];
    q[(size_t)t * Qd + j] = acc;
}

__global__ __launch_bounds__(256) void scores_kernel(const float* __restrict__ q,
                                                     const __bf16* __restrict__ kt,
                                                     float* __restrict__ sc) {
    int s  = blockIdx.x * 256 + threadIdx.x;
    int bn = blockIdx.y;
    int t  = blockIdx.z;
    int b = bn >> 2, n = bn & 3;
    const __bf16* kp = kt + ((size_t)b * HW + s) * Qd + n * 64;
    const float* qp = q + (size_t)t * Qd + n * 64;
    float acc = 0.f;
#pragma unroll 8
    for (int d = 0; d < 64; d++) acc += qp[d] * (float)kp[d];
    sc[(((size_t)t * 4 + b) * 4 + n) * HW + s] = acc * 0.125f;
}

__global__ __launch_bounds__(256) void softmax_kernel(float* __restrict__ sc) {
    float* p = sc + (size_t)blockIdx.x * HW;
    __shared__ float red[256];
    int tid = threadIdx.x;
    float mx = -1e30f;
    for (int s = tid; s < HW; s += 256) mx = fmaxf(mx, p[s]);
    red[tid] = mx; __syncthreads();
    for (int o = 128; o > 0; o >>= 1) { if (tid < o) red[tid] = fmaxf(red[tid], red[tid + o]); __syncthreads(); }
    mx = red[0]; __syncthreads();
    float sum = 0.f;
    for (int s = tid; s < HW; s += 256) { float e = __expf(p[s] - mx); p[s] = e; sum += e; }
    red[tid] = sum; __syncthreads();
    for (int o = 128; o > 0; o >>= 1) { if (tid < o) red[tid] += red[tid + o]; __syncthreads(); }
    float inv = 1.f / red[0];
    for (int s = tid; s < HW; s += 256) p[s] *= inv;
}

__global__ __launch_bounds__(256) void attend_kernel(const float* __restrict__ sc,
                                                     const __bf16* __restrict__ vt,
                                                     float* __restrict__ att) {
    int tbn = blockIdx.x;
    int t = tbn >> 4, b = (tbn >> 2) & 3, n = tbn & 3;
    int tid = threadIdx.x;
    int d = tid & 63, part = tid >> 6;
    const float* p = sc + (size_t)tbn * HW;
    float acc = 0.f;
    for (int s = part * 4096; s < (part + 1) * 4096; s++)
        acc += p[s] * (float)vt[((size_t)b * HW + s) * Qd + n * 64 + d];
    __shared__ float sh[256];
    sh[tid] = acc; __syncthreads();
    if (part == 0) {
        float tot = sh[d] + sh[64 + d] + sh[128 + d] + sh[192 + d];
        att[((size_t)(t * 4 + b)) * Qd + n * 64 + d] = tot;
    }
}

__global__ __launch_bounds__(256) void state_kernel(const float* __restrict__ att,
                                                    const float* __restrict__ w,
                                                    const float* __restrict__ bias,
                                                    float* __restrict__ st) {
    int tb = blockIdx.x, j = threadIdx.x;
    float acc = bias[j];
    const float* a = att + (size_t)tb * Qd;
    const float* wr = w + (size_t)j * Qd;
    for (int i = 0; i < Qd; i++) acc += a[i] * wr[i];
    st[(size_t)tb * Qd + j] = acc;
}

__global__ __launch_bounds__(384) void film_kernel(const float* __restrict__ st,
                                                   const float* __restrict__ w,
                                                   const float* __restrict__ bias,
                                                   float* __restrict__ gamma,
                                                   float* __restrict__ beta) {
    int tb = blockIdx.x, j = threadIdx.x;
    float acc = bias[j];
    const float* s = st + (size_t)tb * Qd;
    const float* wr = w + (size_t)j * Qd;
    for (int i = 0; i < Qd; i++) acc += s[i] * wr[i];
    if (j < Ee) gamma[(size_t)tb * Ee + j] = acc;
    else        beta[(size_t)tb * Ee + (j - Ee)] = acc;
}

// x[img][hw][e] = ev*gate*(1+gamma) + beta   (channels-last bf16, img = t*B+b)
__global__ __launch_bounds__(192) void build_x(const float* __restrict__ ev,
                                               const float* __restrict__ gt,
                                               const float* __restrict__ gamma,
                                               const float* __restrict__ beta,
                                               __bf16* __restrict__ x) {
    int e = threadIdx.x;
    int img = blockIdx.y;
    int hw = blockIdx.x;
    int b = img & 3;
    size_t bi = ((size_t)b * HW + hw) * Ee + e;
    float val = ev[bi] * gt[bi] * (1.f + gamma[(size_t)img * Ee + e]) + beta[(size_t)img * Ee + e];
    x[((size_t)img * HW + hw) * Ee + e] = (__bf16)val;
}

// ---------------------------------------------------------------------------
// 3x3 conv as implicit GEMM, fused BN2+SiLU+mask.
// Block = 192 threads (6 waves) for one (img, h, 32-col strip).
//   - whole 3-row x 34-col x 192-ch input slab staged once in LDS (async),
//     borders zero-padded -> no divergence in the compute loop
//   - each wave: 32 e_out x 32 spatial; weight frags register-prefetched
// ---------------------------------------------------------------------------
__global__ __launch_bounds__(192) void conv3x3_wmma(
    const __bf16* __restrict__ Xc,      // [16][128][128][192] bf16
    const __bf16* __restrict__ Wc,      // [9][192][192] bf16
    const float* __restrict__ s2, const float* __restrict__ h2,
    const unsigned char* __restrict__ mask,   // active_mask (B,S) bool
    float* __restrict__ outp)           // [16][192][128][128] f32
{
    __shared__ __bf16 Xs[3][34][200];   // 192 ch + 8 pad (400B row: no bank conflicts)
    const int tid  = threadIdx.x;
    const int lane = tid & 31;
    const int wave = tid >> 5;          // 0..5 -> e_out 32-tile
    const int half = lane >> 4;
    const int nn   = lane & 15;
    const int w0  = blockIdx.x << 5;    // 32 output columns
    const int h   = blockIdx.y;
    const int img = blockIdx.z;
    const int m0  = wave << 5;

    // --- stage input slab: rows h-1..h+1, cols w0-1..w0+32, all 192 channels ---
    v8bf zero = {};
    for (int idx = tid; idx < 3 * 34 * 24; idx += 192) {
        int q  = idx % 24;
        int cc = (idx / 24) % 34;
        int rr = idx / (24 * 34);
        int hh = h - 1 + rr;
        int ww = w0 - 1 + cc;
        __bf16* dst = &Xs[rr][cc][q * 8];
        if (hh >= 0 && hh < Hh && ww >= 0 && ww < Ww)
            copy16(Xc + (((size_t)img * Hh + hh) * Ww + ww) * Ee + q * 8, dst);
        else
            *reinterpret_cast<v8bf*>(dst) = zero;
    }
    copy_wait_all();

    const __bf16* arow0 = Wc + (size_t)(m0 + nn) * Ee;        // + tap*Ee*Ee
    const __bf16* arow1 = Wc + (size_t)(m0 + 16 + nn) * Ee;

    auto ldA2 = [&](int step, v16bf& A0, v16bf& A1) {
        int tap = step / 6;
        int e0  = (step - tap * 6) * 32;
        size_t toff = (size_t)tap * Ee * Ee;
        A0 = ldA(arow0 + toff, e0, half);
        A1 = ldA(arow1 + toff, e0, half);
    };

    v8f acc[4] = {};                    // [mt*2 + j] : 2 m-tiles x 2 spatial subtiles
    v16bf A0, A1;
    ldA2(0, A0, A1);
    for (int step = 0; step < 54; step++) {
        v16bf nA0 = A0, nA1 = A1;
        if (step + 1 < 54) ldA2(step + 1, nA0, nA1);   // prefetch next weights
        int tap = step / 6;
        int e0  = (step - tap * 6) * 32;
        int kh = tap / 3, kw = tap % 3;
#pragma unroll
        for (int j = 0; j < 2; j++) {
            int c = j * 16 + nn + kw;                  // slab col (zero-padded, no checks)
            v16bf bf = mk16(ldv8(&Xs[kh][c][e0 + half * 8]),
                            ldv8(&Xs[kh][c][e0 + 16 + half * 8]));
            acc[j]     = wmma_bf16(A0, bf, acc[j]);
            acc[2 + j] = wmma_bf16(A1, bf, acc[2 + j]);
        }
        A0 = nA0; A1 = nA1;
    }

    int t = img >> 2, b = img & 3;
    float mk = mask[b * Ss + t] ? 1.f : 0.f;           // active_mask.T[t,b]
#pragma unroll
    for (int mt = 0; mt < 2; mt++) {
#pragma unroll
        for (int j = 0; j < 2; j++) {
            int col = w0 + j * 16 + nn;
#pragma unroll
            for (int r = 0; r < 8; r++) {
                int m = m0 + mt * 16 + r + half * 8;
                float y = acc[mt * 2 + j][r] * s2[m] + h2[m];
                y = y / (1.f + __expf(-y));            // SiLU
                outp[(((size_t)img * Ee + m) * Hh + h) * Ww + col] = y * mk;
            }
        }
    }
}

// ---------------------------------------------------------------------------
// Orchestration
// ---------------------------------------------------------------------------
extern "C" void kernel_launch(void* const* d_in, const int* in_sizes, int n_in,
                              void* d_out, int out_size, void* d_ws, size_t ws_size,
                              hipStream_t stream) {
    (void)in_sizes; (void)n_in; (void)out_size; (void)ws_size;

    const float* memory   = (const float*)d_in[0];
    const unsigned char* active_mask = (const unsigned char*)d_in[1];
    const float* stem_q   = (const float*)d_in[2];
    const float* wproj_f  = (const float*)d_in[3];
    const float* inw      = (const float*)d_in[4];
    const float* inb      = (const float*)d_in[5];
    const float* outw     = (const float*)d_in[6];
    const float* outb     = (const float*)d_in[7];
    const float* evw_f    = (const float*)d_in[8];
    const float* bn1g = (const float*)d_in[9],  *bn1b = (const float*)d_in[10];
    const float* bn1m = (const float*)d_in[11], *bn1v = (const float*)d_in[12];
    const float* gatew_f  = (const float*)d_in[13];
    const float* gateb    = (const float*)d_in[14];
    const float* filmw    = (const float*)d_in[15];
    const float* filmb    = (const float*)d_in[16];
    const float* convw_f  = (const float*)d_in[17];
    const float* bn2g = (const float*)d_in[18], *bn2b = (const float*)d_in[19];
    const float* bn2m = (const float*)d_in[20], *bn2v = (const float*)d_in[21];

    char* ws = (char*)d_ws;
    size_t off = 0;
    auto alloc = [&](size_t bytes) { size_t r = off; off = (off + bytes + 255) & ~(size_t)255; return r; };

    __bf16* memt  = (__bf16*)(ws + alloc((size_t)Bn * HW * Cc * 2));
    __bf16* wproj = (__bf16*)(ws + alloc((size_t)Qd * Cc * 2));
    __bf16* wkbf  = (__bf16*)(ws + alloc((size_t)Qd * Qd * 2));
    __bf16* wvbf  = (__bf16*)(ws + alloc((size_t)Qd * Qd * 2));
    __bf16* wevbf = (__bf16*)(ws + alloc((size_t)Ee * Cc * 2));
    __bf16* wgtbf = (__bf16*)(ws + alloc((size_t)Ee * Qd * 2));
    __bf16* wconv = (__bf16*)(ws + alloc((size_t)9 * Ee * Ee * 2));
    __bf16* mt_t  = (__bf16*)(ws + alloc((size_t)Bn * HW * Qd * 2));
    __bf16* k_t   = (__bf16*)(ws + alloc((size_t)Bn * HW * Qd * 2));
    __bf16* v_t   = (__bf16*)(ws + alloc((size_t)Bn * HW * Qd * 2));
    float*  evb   = (float*)(ws + alloc((size_t)Bn * HW * Ee * 4));
    float*  gtb   = (float*)(ws + alloc((size_t)Bn * HW * Ee * 4));
    __bf16* xbuf  = (__bf16*)(ws + alloc((size_t)Ss * Bn * HW * Ee * 2));
    float*  sc    = (float*)(ws + alloc((size_t)Ss * Bn * 4 * HW * 4));
    float*  qbuf  = (float*)(ws + alloc((size_t)Ss * Qd * 4));
    float*  att   = (float*)(ws + alloc((size_t)Ss * Bn * Qd * 4));
    float*  st    = (float*)(ws + alloc((size_t)Ss * Bn * Qd * 4));
    float*  gamma = (float*)(ws + alloc((size_t)Ss * Bn * Ee * 4));
    float*  beta  = (float*)(ws + alloc((size_t)Ss * Bn * Ee * 4));
    float*  s1    = (float*)(ws + alloc(Ee * 4));
    float*  h1    = (float*)(ws + alloc(Ee * 4));
    float*  s2    = (float*)(ws + alloc(Ee * 4));
    float*  h2    = (float*)(ws + alloc(Ee * 4));

    // --- prep ---
    prep_bn<<<1, 192, 0, stream>>>(bn1g, bn1b, bn1m, bn1v, bn2g, bn2b, bn2m, bn2v, s1, h1, s2, h2);
    cvt_f32_bf16<<<(Qd * Cc + 255) / 256, 256, 0, stream>>>(wproj_f, wproj, Qd * Cc);
    cvt_f32_bf16<<<(Qd * Qd + 255) / 256, 256, 0, stream>>>(inw + (size_t)Qd * Qd, wkbf, Qd * Qd);
    cvt_f32_bf16<<<(Qd * Qd + 255) / 256, 256, 0, stream>>>(inw + (size_t)2 * Qd * Qd, wvbf, Qd * Qd);
    cvt_f32_bf16<<<(Ee * Cc + 255) / 256, 256, 0, stream>>>(evw_f, wevbf, Ee * Cc);
    cvt_f32_bf16<<<(Ee * Qd + 255) / 256, 256, 0, stream>>>(gatew_f, wgtbf, Ee * Qd);
    prep_wconv<<<(9 * Ee * Ee + 255) / 256, 256, 0, stream>>>(convw_f, wconv, 9 * Ee * Ee);
    transpose_mem<<<dim3(HW / 32, Cc / 32, Bn), 256, 0, stream>>>(memory, memt);

    // --- WMMA GEMMs: grid = (N/64, 1, B), 8 waves/block, M covered by waves ---
    dim3 gg(HW / 64, 1, Bn);
    gemm_bf16_wmma<<<gg, 256, 0, stream>>>(wproj, memt, mt_t, nullptr, nullptr, Qd, HW, Cc, 0);
    gemm_bf16_wmma<<<gg, 256, 0, stream>>>(wkbf, mt_t, k_t, inb + Qd, nullptr, Qd, HW, Qd, 0);
    gemm_bf16_wmma<<<gg, 256, 0, stream>>>(wvbf, mt_t, v_t, inb + 2 * Qd, nullptr, Qd, HW, Qd, 0);
    gemm_bf16_wmma<<<gg, 256, 0, stream>>>(wevbf, memt, evb, s1, h1, Ee, HW, Cc, 1);
    gemm_bf16_wmma<<<gg, 256, 0, stream>>>(wgtbf, mt_t, gtb, gateb, nullptr, Ee, HW, Qd, 2);

    // --- attention (tiny) ---
    qproj_kernel<<<Ss, 256, 0, stream>>>(stem_q, inw, inb, qbuf);
    scores_kernel<<<dim3(HW / 256, Bn * 4, Ss), 256, 0, stream>>>(qbuf, k_t, sc);
    softmax_kernel<<<Ss * Bn * 4, 256, 0, stream>>>(sc);
    attend_kernel<<<Ss * Bn * 4, 256, 0, stream>>>(sc, v_t, att);
    state_kernel<<<Ss * Bn, 256, 0, stream>>>(att, outw, outb, st);
    film_kernel<<<Ss * Bn, 384, 0, stream>>>(st, filmw, filmb, gamma, beta);

    // --- FiLM-modulated activation tensor (channels-last bf16) ---
    build_x<<<dim3(HW, Ss * Bn), 192, 0, stream>>>(evb, gtb, gamma, beta, xbuf);

    // --- 3x3 conv: LDS-staged implicit WMMA GEMM + BN2 + SiLU + mask ---
    conv3x3_wmma<<<dim3(Ww / 32, Hh, Ss * Bn), 192, 0, stream>>>(
        xbuf, wconv, s2, h2, active_mask, (float*)d_out);
}
